// SharedBasis_75024488727155
// MI455X (gfx1250) — matrix operand order repaired and checked by reading, not compiled, not run on hardware
//
#include <hip/hip_runtime.h>

// ---------------------------------------------------------------------------
// SharedBasis pipeline for MI455X (gfx1250, wave32).
//  - All feature GEMMs run through a bf16 WMMA kernel (v_wmma_f32_16x16x32_bf16),
//    one 16x16 C tile per wave, f32 accumulate. The per-block B (weight) slab is
//    staged into LDS once via global_load_async_to_lds_b128 (ASYNCcnt path) and
//    read from LDS one bf16 element per k, cutting global weight traffic 8x.
//  - Laplacian + parallel cyclic Jacobi eigensolver in f32 (67MB working set,
//    fits in the 192MB L2, so Jacobi sweeps run at L2 bandwidth).
// ---------------------------------------------------------------------------

typedef __attribute__((ext_vector_type(16))) __bf16 v16bf;
typedef __attribute__((ext_vector_type(8)))  float  v8f;

#define KNN_K 16

__device__ __forceinline__ __bf16 f32_to_bf16(float f) {
  unsigned u = __builtin_bit_cast(unsigned, f);
  unsigned r = u + 0x7FFFu + ((u >> 16) & 1u);   // round-to-nearest-even
  unsigned short h = (unsigned short)(r >> 16);
  return __builtin_bit_cast(__bf16, h);
}

__device__ __forceinline__ float leaky02(float v) { return v > 0.f ? v : 0.2f * v; }

// ---------------------------------------------------------------------------
// WMMA GEMM: C[M,N] = A[M,Kp] @ B[Kp,N]  (bf16 in, f32 out)
// epilogue: 0 = none, 1 = bias + leaky_relu(0.2), 2 = bias only.
// Block = 8 waves = 8 consecutive M-tiles sharing one N-tile. The Kp x 16 B
// slab is async-copied to LDS once per block (ASYNCcnt), then each wave's B
// fragment comes from LDS. M is always a multiple of 128 so there are no
// partial blocks and EXEC is all-ones at every v_wmma (ISA requirement).
// Fragment layouts follow CDNA5 ISA 7.12.2 (16-bit A 16x32, B 32x16).
// ---------------------------------------------------------------------------
__global__ void sb_gemm(const __bf16* __restrict__ A, const __bf16* __restrict__ Bm,
                        const float* __restrict__ bias, float* __restrict__ C,
                        int M, int N, int Kp, int epi) {
  __shared__ __attribute__((aligned(16))) __bf16 Bsh[384 * 16];   // 12 KB max slab
  const int tilesMB = M >> 7;                 // blocks along M (8 tiles each)
  const int tmb     = blockIdx.x % tilesMB;
  const int tile_n  = blockIdx.x / tilesMB;
  const int wv   = threadIdx.x >> 5;
  const int lane = threadIdx.x & 31;
  const int tile_m = tmb * 8 + wv;

  // ---- async-stage B slab [Kp x 16] for this tile_n into LDS ----
  {
    const int nchunks = Kp * 2;               // 16B chunks: 2 per B row
    for (int r = threadIdx.x; r < nchunks; r += 256) {
      int k = r >> 1, hf = r & 1;
      const __bf16* g = Bm + (long long)k * N + tile_n * 16 + hf * 8;
      unsigned loff = (unsigned)(size_t)(&Bsh[k * 16 + hf * 8]);  // low 32 bits give the LDS address
      asm volatile("global_load_async_to_lds_b128 %0, %1, off"
                   :: "v"(loff), "v"(g) : "memory");
    }
    asm volatile("s_wait_asynccnt 0x0" ::: "memory");
    __syncthreads();
  }

  v8f acc = {};
  const int rowA = tile_m * 16 + (lane & 15);
  const int kadd = (lane >> 4) * 8;                  // A half-lane K offset
  const int kB   = (lane & 15);                      // B column held by this lane

  #pragma unroll 2
  for (int k0 = 0; k0 < Kp; k0 += 32) {
    union { unsigned u[8]; v16bf v; } af, bf;
    const __bf16* Arow = A + (long long)rowA * Kp + k0 + kadd;
    __builtin_prefetch(Arow + 32, 0, 1);             // global_prefetch next k-slab
    #pragma unroll
    for (int vg = 0; vg < 8; ++vg) {
      int kk = (vg < 4) ? (2 * vg) : (16 + 2 * (vg - 4));
      af.u[vg] = *(const unsigned*)(Arow + kk);      // 2x global_load_b128
    }
    #pragma unroll
    for (int s = 0; s < 16; ++s) {
      int kk = (s < 8) ? (kadd + s) : (16 + kadd + (s - 8));
      bf.v[s] = Bsh[(k0 + kk) * 16 + kB];
    }
    acc = __builtin_amdgcn_wmma_f32_16x16x32_bf16(
        false, af.v, false, bf.v, (short)0, acc, false, false);
  }

  int n     = tile_n * 16 + (lane & 15);
  int mbase = tile_m * 16 + (lane >> 4) * 8;
  float bv = (epi != 0 && bias) ? bias[n] : 0.f;
  #pragma unroll
  for (int j = 0; j < 8; ++j) {
    float v = acc[j] + bv;
    if (epi == 1) v = leaky02(v);
    C[(long long)(mbase + j) * N + n] = v;
  }
}

// ---------------------------------------------------------------------------
// Elementwise / staging kernels
// ---------------------------------------------------------------------------
__global__ void sb_cvt_pad(const float* __restrict__ s, __bf16* __restrict__ d,
                           int rows, int cols, int rp, int cp) {
  int t = blockIdx.x * blockDim.x + threadIdx.x;
  if (t >= rp * cp) return;
  int c = t % cp, r = t / cp;
  float v = (r < rows && c < cols) ? s[(long long)r * cols + c] : 0.f;
  d[t] = f32_to_bf16(v);
}

__global__ void sb_initfeat(float* __restrict__ f, int P) {
  int t = blockIdx.x * blockDim.x + threadIdx.x;
  if (t >= P * 4) return;
  f[t] = ((t & 3) < 3) ? 1.f : (float)KNN_K;
}

__global__ void sb_build_h(const float* __restrict__ x, const float* __restrict__ pos,
                           const int* __restrict__ idx, __bf16* __restrict__ h,
                           int Bb, int Np, int Cin, int Hp) {
  long long t = blockIdx.x * (long long)blockDim.x + threadIdx.x;
  long long total = (long long)Bb * Np * KNN_K * Hp;
  if (t >= total) return;
  int c = (int)(t % Hp);
  long long r = t / Hp;
  int k = (int)(r % KNN_K);
  long long pi = r / KNN_K;     // b*Np+i
  int i = (int)(pi % Np);
  int b = (int)(pi / Np);
  int j = idx[pi * KNN_K + k];
  const float* pb = pos + (long long)b * Np * 3;
  float v = 0.f;
  if (c < Cin) {
    v = x[((long long)b * Np + j) * Cin + c];
  } else if (c < Cin + 3) {
    int d = c - Cin;
    v = pb[i * 3 + d] - pb[j * 3 + d];
  } else if (c == Cin + 3) {
    float s = 0.f;
    for (int d = 0; d < 3; ++d) { float rd = pb[i*3+d] - pb[j*3+d]; s += rd * rd; }
    v = s;
  }
  h[r * Hp + c] = f32_to_bf16(v);
}

__global__ void sb_combine(const float* __restrict__ selfo, const float* __restrict__ edgeo,
                           float* __restrict__ out, int P, int C) {
  int t = blockIdx.x * blockDim.x + threadIdx.x;
  if (t >= P * C) return;
  int c = t % C; int p = t / C;
  float m = 0.f;
  for (int k = 0; k < KNN_K; ++k) m += edgeo[((long long)p * KNN_K + k) * C + c];
  out[t] = leaky02(selfo[t] + m * (1.f / KNN_K));
}

__global__ void sb_concat_bf16(const float* __restrict__ a, int Ca,
                               const float* __restrict__ b, int Cb,
                               __bf16* __restrict__ d, int P, int Cp) {
  int t = blockIdx.x * blockDim.x + threadIdx.x;
  if (t >= P * Cp) return;
  int c = t % Cp; int p = t / Cp;
  float v = 0.f;
  if (c < Ca)            v = a[(long long)p * Ca + c];
  else if (c < Ca + Cb)  v = b[(long long)p * Cb + (c - Ca)];
  d[t] = f32_to_bf16(v);
}

// ---------------------------------------------------------------------------
// KNN / FPS / gathers / interpolation
// ---------------------------------------------------------------------------
__global__ void sb_knn(const float* __restrict__ pos, int* __restrict__ idx, int Bb, int Np) {
  int t = blockIdx.x * blockDim.x + threadIdx.x;
  if (t >= Bb * Np) return;
  int b = t / Np, i = t % Np;
  const float* pb = pos + (long long)b * Np * 3;
  float xi = pb[i*3], yi = pb[i*3+1], zi = pb[i*3+2];
  float bd[KNN_K]; int bi[KNN_K];
  for (int s = 0; s < KNN_K; ++s) { bd[s] = 3.4e38f; bi[s] = 0; }
  for (int j = 0; j < Np; ++j) {
    float dx = pb[j*3]-xi, dy = pb[j*3+1]-yi, dz = pb[j*3+2]-zi;
    float d = dx*dx + dy*dy + dz*dz;
    if (d < bd[KNN_K-1]) {
      int s = KNN_K - 1;
      while (s > 0 && bd[s-1] > d) { bd[s] = bd[s-1]; bi[s] = bi[s-1]; --s; }
      bd[s] = d; bi[s] = j;
    }
  }
  for (int s = 0; s < KNN_K; ++s) idx[(long long)t * KNN_K + s] = bi[s];
}

__global__ void sb_fps(const float* __restrict__ pos, int* __restrict__ out, int Np, int npoint) {
  __shared__ float dist[1024];
  __shared__ float rv[256];
  __shared__ int   ri[256];
  __shared__ int   s_last;
  int b = blockIdx.x, t = threadIdx.x;
  const float* pb = pos + (long long)b * Np * 3;
  for (int i = t; i < Np; i += 256) dist[i] = 1e10f;
  if (t == 0) { out[(long long)b * npoint] = 0; s_last = 0; }
  __syncthreads();
  for (int s = 1; s < npoint; ++s) {
    int last = s_last;
    float lx = pb[last*3], ly = pb[last*3+1], lz = pb[last*3+2];
    float bm = -1.f; int ba = 0;
    for (int i = t; i < Np; i += 256) {
      float dx = pb[i*3]-lx, dy = pb[i*3+1]-ly, dz = pb[i*3+2]-lz;
      float nd = fminf(dist[i], dx*dx + dy*dy + dz*dz);
      dist[i] = nd;
      if (nd > bm) { bm = nd; ba = i; }
    }
    rv[t] = bm; ri[t] = ba;
    __syncthreads();
    for (int off = 128; off > 0; off >>= 1) {
      if (t < off && rv[t+off] > rv[t]) { rv[t] = rv[t+off]; ri[t] = ri[t+off]; }
      __syncthreads();
    }
    if (t == 0) { s_last = ri[0]; out[(long long)b * npoint + s] = ri[0]; }
    __syncthreads();
  }
}

__global__ void sb_gather(const float* __restrict__ src, const int* __restrict__ gi,
                          float* __restrict__ dst, int Bb, int Nout, int Nin, int C) {
  long long t = blockIdx.x * (long long)blockDim.x + threadIdx.x;
  long long total = (long long)Bb * Nout * C;
  if (t >= total) return;
  int c = (int)(t % C);
  long long p = t / C;
  int i = (int)(p % Nout);
  int b = (int)(p / Nout);
  int j = gi[(long long)b * Nout + i];
  dst[t] = src[((long long)b * Nin + j) * C + c];
}

__global__ void sb_three_nn(const float* __restrict__ pu, const float* __restrict__ pk,
                            int* __restrict__ i3, float* __restrict__ w3,
                            int Bb, int Nu, int Nk) {
  int t = blockIdx.x * blockDim.x + threadIdx.x;
  if (t >= Bb * Nu) return;
  int b = t / Nu, i = t % Nu;
  const float* a = pu + ((long long)b * Nu + i) * 3;
  const float* kb = pk + (long long)b * Nk * 3;
  float d0 = 3e38f, d1 = 3e38f, d2 = 3e38f; int j0 = 0, j1 = 0, j2 = 0;
  for (int j = 0; j < Nk; ++j) {
    float dx = a[0]-kb[j*3], dy = a[1]-kb[j*3+1], dz = a[2]-kb[j*3+2];
    float d = dx*dx + dy*dy + dz*dz;
    if (d < d0)      { d2=d1; j2=j1; d1=d0; j1=j0; d0=d; j0=j; }
    else if (d < d1) { d2=d1; j2=j1; d1=d; j1=j; }
    else if (d < d2) { d2=d; j2=j; }
  }
  float r0 = 1.f/(d0+1e-8f), r1 = 1.f/(d1+1e-8f), r2 = 1.f/(d2+1e-8f);
  float s = r0 + r1 + r2;
  i3[(long long)t*3+0] = j0; i3[(long long)t*3+1] = j1; i3[(long long)t*3+2] = j2;
  w3[(long long)t*3+0] = r0/s; w3[(long long)t*3+1] = r1/s; w3[(long long)t*3+2] = r2/s;
}

__global__ void sb_interp(const float* __restrict__ fk, const int* __restrict__ i3,
                          const float* __restrict__ w3, float* __restrict__ out,
                          int Bb, int Nu, int Nk, int C) {
  long long t = blockIdx.x * (long long)blockDim.x + threadIdx.x;
  long long total = (long long)Bb * Nu * C;
  if (t >= total) return;
  int c = (int)(t % C);
  long long p = t / C;
  int b = (int)(p / Nu);
  float s = 0.f;
  for (int k = 0; k < 3; ++k) {
    int j = i3[p*3+k];
    s += w3[p*3+k] * fk[((long long)b * Nk + j) * C + c];
  }
  out[t] = s;
}

// ---------------------------------------------------------------------------
// latent, normals, edge weights, loss
// ---------------------------------------------------------------------------
__global__ void sb_l2norm(const float* __restrict__ in, float* __restrict__ out, int P, int C) {
  int t = blockIdx.x * blockDim.x + threadIdx.x;
  if (t >= P) return;
  const float* r = in + (long long)t * C;
  float s = 0.f;
  for (int c = 0; c < C; ++c) s += r[c] * r[c];
  float inv = 1.f / fmaxf(sqrtf(s), 1e-8f);
  for (int c = 0; c < C; ++c) out[(long long)t * C + c] = r[c] * inv;
}

__global__ void sb_normals(const float* __restrict__ ps, const int* __restrict__ ci,
                           float* __restrict__ nrm, int Bb, int Nn, int HW) {
  int t = blockIdx.x * blockDim.x + threadIdx.x;
  if (t >= Bb * Nn) return;
  int b = t / Nn;
  const float mn[3] = {0.485f, 0.456f, 0.406f};
  const float sd[3] = {0.229f, 0.224f, 0.225f};
  int c0 = ci[t];
  float f[3];
  for (int c = 0; c < 3; ++c) {
    float v = ps[((long long)b * 3 + c) * HW + c0];
    v = (v * sd[c] + mn[c]) * 2.f - 1.f;
    f[c] = fminf(fmaxf(v, -1.f), 1.f);
  }
  float nx = f[1], ny = f[2];
  float nz = sqrtf(fmaxf(1.f - nx*nx - ny*ny, 0.f) + 1e-8f);
  float inv = 1.f / fmaxf(sqrtf(nx*nx + ny*ny + nz*nz), 1e-8f);
  nrm[(long long)t*3+0] = nx * inv;
  nrm[(long long)t*3+1] = ny * inv;
  nrm[(long long)t*3+2] = nz * inv;
}

__global__ void sb_wval(const float* __restrict__ lat, const int* __restrict__ idx,
                        float* __restrict__ Wv, int Bb, int Nn, int C) {
  int t = blockIdx.x * blockDim.x + threadIdx.x;
  if (t >= Bb * Nn * KNN_K) return;
  long long pi = t / KNN_K;
  int b = (int)(pi / Nn);
  int j = idx[t];
  const float* li = lat + pi * C;
  const float* lj = lat + ((long long)b * Nn + j) * C;
  float s = 0.f;
  for (int c = 0; c < C; ++c) { float d = lj[c] - li[c]; s += d * d; }
  Wv[t] = fmaxf(s, 0.f);
}

__global__ void sb_zero(float* p, int n) {
  int t = blockIdx.x * blockDim.x + threadIdx.x;
  if (t < n) p[t] = 0.f;
}

__global__ void sb_idx_copy(float* __restrict__ out, const int* __restrict__ idx, int n) {
  int t = blockIdx.x * blockDim.x + threadIdx.x;
  if (t < n) out[t] = (float)idx[t];
}

__global__ void sb_loss(const float* __restrict__ Wv, const float* __restrict__ nrm,
                        const int* __restrict__ idx, const float* __restrict__ alpha,
                        float* __restrict__ loss, int Bb, int Nn) {
  __shared__ float red[256];
  int t = blockIdx.x * blockDim.x + threadIdx.x;
  int total = Bb * Nn * KNN_K;
  float val = 0.f;
  if (t < total) {
    long long pi = t / KNN_K;
    int b = (int)(pi / Nn);
    int j = idx[t];
    const float* ni = nrm + pi * 3;
    const float* nj = nrm + ((long long)b * Nn + j) * 3;
    float d = 0.f;
    for (int c = 0; c < 3; ++c) { float r = nj[c] - ni[c]; d += r * r; }
    float e = Wv[t] - alpha[0] * d;
    val = e * e;
  }
  red[threadIdx.x] = val;
  __syncthreads();
  for (int off = 128; off > 0; off >>= 1) {
    if (threadIdx.x < off) red[threadIdx.x] += red[threadIdx.x + off];
    __syncthreads();
  }
  if (threadIdx.x == 0) atomicAdd(loss, red[0] / (float)total);
}

// ---------------------------------------------------------------------------
// Laplacian + parallel cyclic Jacobi eigensolver
// ---------------------------------------------------------------------------
__global__ void sb_scatter(const float* __restrict__ Wv, const int* __restrict__ idx,
                           float* __restrict__ A, int Bb, int Nn) {
  int t = blockIdx.x * blockDim.x + threadIdx.x;
  if (t >= Bb * Nn * KNN_K) return;
  long long pi = t / KNN_K;
  int i = (int)(pi % Nn);
  int b = (int)(pi / Nn);
  int j = idx[t];
  A[((long long)b * Nn + i) * Nn + j] = Wv[t];
}

__global__ void sb_degree(const float* __restrict__ A, float* __restrict__ dis, int Bb, int Nn) {
  int t = blockIdx.x * blockDim.x + threadIdx.x;
  if (t >= Bb * Nn) return;
  int i = t % Nn, b = t / Nn;
  const float* Ab = A + (long long)b * Nn * Nn;
  float s = 0.f;
  for (int j = 0; j < Nn; ++j)
    s += 0.5f * (Ab[(long long)i * Nn + j] + Ab[(long long)j * Nn + i]);
  dis[t] = rsqrtf(s + 1e-6f);
}

__global__ void sb_buildL(const float* __restrict__ A, const float* __restrict__ dis,
                          float* __restrict__ L, int Bb, int Nn) {
  long long t = blockIdx.x * (long long)blockDim.x + threadIdx.x;
  long long total = (long long)Bb * Nn * Nn;
  if (t >= total) return;
  int j = (int)(t % Nn);
  long long r = t / Nn;
  int i = (int)(r % Nn);
  int b = (int)(r / Nn);
  const float* Ab = A + (long long)b * Nn * Nn;
  float as = 0.5f * (Ab[(long long)i * Nn + j] + Ab[(long long)j * Nn + i]);
  float v = -dis[(long long)b * Nn + i] * as * dis[(long long)b * Nn + j];
  if (i == j) v += 1.f + 1e-5f;
  L[t] = v;
}

__global__ void sb_initU(float* __restrict__ U, int Bb, int Nn) {
  long long t = blockIdx.x * (long long)blockDim.x + threadIdx.x;
  long long total = (long long)Bb * Nn * Nn;
  if (t >= total) return;
  int j = (int)(t % Nn);
  int i = (int)((t / Nn) % Nn);
  U[t] = (i == j) ? 1.f : 0.f;
}

__device__ __forceinline__ void sb_rr(int step, int t, int n, int& p, int& q) {
  // round-robin (chess tournament) disjoint pairing
  int a  = (t == 0) ? 0 : (1 + ((t - 1 + step) % (n - 1)));
  int bp = n - 1 - t;
  int bb = 1 + ((bp - 1 + step) % (n - 1));
  p = a < bb ? a : bb;
  q = a < bb ? bb : a;
}

__global__ void sb_jrot(const float* __restrict__ L, float* __restrict__ cs,
                        int step, int Bb, int Nn) {
  int t = blockIdx.x * blockDim.x + threadIdx.x;
  int P2 = Nn >> 1;
  if (t >= Bb * P2) return;
  int b = t / P2, pr = t % P2;
  int p, q; sb_rr(step, pr, Nn, p, q);
  const float* Lb = L + (long long)b * Nn * Nn;
  float app = Lb[(long long)p * Nn + p];
  float aqq = Lb[(long long)q * Nn + q];
  float apq = Lb[(long long)p * Nn + q];
  float c = 1.f, s = 0.f;
  if (fabsf(apq) > 1e-20f) {
    float tau = (aqq - app) / (2.f * apq);
    float tt = (tau >= 0.f ? 1.f : -1.f) / (fabsf(tau) + sqrtf(1.f + tau * tau));
    c = rsqrtf(1.f + tt * tt);
    s = tt * c;
  }
  cs[(long long)t * 2 + 0] = c;
  cs[(long long)t * 2 + 1] = s;
}

__global__ void sb_jrows(float* __restrict__ L, const float* __restrict__ cs,
                         int step, int Bb, int Nn) {
  long long t = blockIdx.x * (long long)blockDim.x + threadIdx.x;
  int P2 = Nn >> 1;
  long long total = (long long)Bb * P2 * Nn;
  if (t >= total) return;
  int col = (int)(t % Nn);
  long long r = t / Nn;
  int pr = (int)(r % P2);
  int b  = (int)(r / P2);
  int p, q; sb_rr(step, pr, Nn, p, q);
  float c = cs[r * 2], s = cs[r * 2 + 1];
  float* Lb = L + (long long)b * Nn * Nn;
  float ap = Lb[(long long)p * Nn + col];
  float aq = Lb[(long long)q * Nn + col];
  Lb[(long long)p * Nn + col] = c * ap - s * aq;
  Lb[(long long)q * Nn + col] = s * ap + c * aq;
}

__global__ void sb_jcols(float* __restrict__ L, float* __restrict__ U,
                         const float* __restrict__ cs, int step, int Bb, int Nn) {
  long long t = blockIdx.x * (long long)blockDim.x + threadIdx.x;
  int P2 = Nn >> 1;
  long long total = (long long)Bb * P2 * Nn;
  if (t >= total) return;
  int row = (int)(t % Nn);
  long long r = t / Nn;
  int pr = (int)(r % P2);
  int b  = (int)(r / P2);
  int p, q; sb_rr(step, pr, Nn, p, q);
  float c = cs[r * 2], s = cs[r * 2 + 1];
  float* Lb = L + (long long)b * Nn * Nn;
  float* Ub = U + (long long)b * Nn * Nn;
  float ap = Lb[(long long)row * Nn + p];
  float aq = Lb[(long long)row * Nn + q];
  Lb[(long long)row * Nn + p] = c * ap - s * aq;
  Lb[(long long)row * Nn + q] = s * ap + c * aq;
  float up = Ub[(long long)row * Nn + p];
  float uq = Ub[(long long)row * Nn + q];
  Ub[(long long)row * Nn + p] = c * up - s * uq;
  Ub[(long long)row * Nn + q] = s * up + c * uq;
}

__global__ void sb_sortperm(const float* __restrict__ L, int* __restrict__ perm, int Nn) {
  __shared__ float key[1024];
  __shared__ int   pm[1024];
  int t = threadIdx.x, b = blockIdx.x;
  key[t] = L[((long long)b * Nn + t) * Nn + t];
  pm[t] = t;
  __syncthreads();
  for (int it = 0; it < Nn; ++it) {
    if (((t ^ it) & 1) == 0 && t + 1 < Nn) {
      if (key[t] > key[t + 1]) {
        float kk = key[t]; key[t] = key[t + 1]; key[t + 1] = kk;
        int pp = pm[t]; pm[t] = pm[t + 1]; pm[t + 1] = pp;
      }
    }
    __syncthreads();
  }
  perm[(long long)b * Nn + t] = pm[t];
}

__global__ void sb_permU(const float* __restrict__ U, const int* __restrict__ perm,
                         float* __restrict__ out, int Bb, int Nn) {
  long long t = blockIdx.x * (long long)blockDim.x + threadIdx.x;
  long long total = (long long)Bb * Nn * Nn;
  if (t >= total) return;
  int j = (int)(t % Nn);
  long long r = t / Nn;
  int i = (int)(r % Nn);
  int b = (int)(r / Nn);
  int pj = perm[(long long)b * Nn + j];
  out[t] = U[((long long)b * Nn + i) * Nn + pj];
}

// ---------------------------------------------------------------------------
// Host orchestration
// ---------------------------------------------------------------------------
static inline int pad32i(int x) { return (x + 31) & ~31; }
static inline int nblk(long long n) { return (int)((n + 255) / 256); }
static inline size_t al256(size_t x) { return (x + 255) & ~(size_t)255; }

static void launch_gemm(hipStream_t st, const __bf16* A, const __bf16* Bm,
                        const float* bias, float* C, int M, int N, int Kp, int epi) {
  // M is always a multiple of 128 here; 8 waves (tiles) per block, exact tiling.
  int blocks = (M >> 7) * (N >> 4);
  sb_gemm<<<blocks, 256, 0, st>>>(A, Bm, bias, C, M, N, Kp, epi);
}

static void run_edge_conv(hipStream_t st, const float* x, const float* pos, const int* idx,
                          const float* w_self, const float* w_edge, float* out,
                          int Bb, int Np, int Cin, int Cout,
                          __bf16* bfA, __bf16* bfB, float* tmpE, float* tmpS) {
  int P = Bb * Np;
  long long RE = (long long)P * KNN_K;
  int Hc = Cin + 4, Hp = pad32i(Hc);
  int Sp = pad32i(Cin);
  // edge branch: h = [x_j, rel, |rel|^2]  -> mean_k (h @ w_edge)
  sb_build_h<<<nblk(RE * Hp), 256, 0, st>>>(x, pos, idx, bfA, Bb, Np, Cin, Hp);
  sb_cvt_pad<<<nblk((long long)Hp * Cout), 256, 0, st>>>(w_edge, bfB, Hc, Cout, Hp, Cout);
  launch_gemm(st, bfA, bfB, nullptr, tmpE, (int)RE, Cout, Hp, 0);
  // self branch: x @ w_self
  sb_cvt_pad<<<nblk((long long)P * Sp), 256, 0, st>>>(x, bfA, P, Cin, P, Sp);
  sb_cvt_pad<<<nblk((long long)Sp * Cout), 256, 0, st>>>(w_self, bfB, Cin, Cout, Sp, Cout);
  launch_gemm(st, bfA, bfB, nullptr, tmpS, P, Cout, Sp, 0);
  sb_combine<<<nblk((long long)P * Cout), 256, 0, st>>>(tmpS, tmpE, out, P, Cout);
}

extern "C" void kernel_launch(void* const* d_in, const int* in_sizes, int n_in,
                              void* d_out, int out_size, void* d_ws, size_t ws_size,
                              hipStream_t stream) {
  (void)in_sizes; (void)n_in; (void)out_size; (void)ws_size;
  const int B = 8, N = 1024, Hh = 224, Ww = 224;
  const int N1 = N / 2, N2 = N / 4;

  const float* xyz   = (const float*)d_in[0];
  const float* ps    = (const float*)d_in[1];
  const int*   cidx  = (const int*)d_in[2];
  const float* w_self0 = (const float*)d_in[3];
  const float* w_edge0 = (const float*)d_in[4];
  const float* w_self1 = (const float*)d_in[5];
  const float* w_edge1 = (const float*)d_in[6];
  const float* w_self2 = (const float*)d_in[7];
  const float* w_edge2 = (const float*)d_in[8];
  const float* w_up1 = (const float*)d_in[9];
  const float* b_up1 = (const float*)d_in[10];
  const float* w_cu1 = (const float*)d_in[11];
  const float* w_edge_cu1 = (const float*)d_in[12];
  const float* w_up0 = (const float*)d_in[13];
  const float* b_up0 = (const float*)d_in[14];
  const float* w_cu0 = (const float*)d_in[15];
  const float* w_edge_cu0 = (const float*)d_in[16];
  const float* w_out = (const float*)d_in[17];
  const float* b_out = (const float*)d_in[18];
  const float* alpha = (const float*)d_in[19];

  float* out = (float*)d_out;
  const long long UN = (long long)B * N * N;
  float* out_U    = out;
  float* out_loss = out + UN;
  float* out_idx  = out + UN + 1;

  // ---- workspace carve ----
  char* wsp = (char*)d_ws;
  auto alloc = [&](size_t bytes) -> void* { void* p = wsp; wsp += al256(bytes); return p; };
  int* i_idx0 = (int*)alloc((size_t)B * N  * KNN_K * 4);
  int* i_idx1 = (int*)alloc((size_t)B * N1 * KNN_K * 4);
  int* i_idx2 = (int*)alloc((size_t)B * N2 * KNN_K * 4);
  int* i_fps1 = (int*)alloc((size_t)B * N1 * 4);
  int* i_fps2 = (int*)alloc((size_t)B * N2 * 4);
  int* i_t3   = (int*)alloc((size_t)B * N * 3 * 4);
  float* f_w3    = (float*)alloc((size_t)B * N * 3 * 4);
  float* f_feat  = (float*)alloc((size_t)B * N * 4 * 4);
  float* f_feat0 = (float*)alloc((size_t)B * N * 64 * 4);
  float* f_pos1  = (float*)alloc((size_t)B * N1 * 3 * 4);
  float* f_f0g   = (float*)alloc((size_t)B * N1 * 64 * 4);
  float* f_feat1 = (float*)alloc((size_t)B * N1 * 128 * 4);
  float* f_pos2  = (float*)alloc((size_t)B * N2 * 3 * 4);
  float* f_f1g   = (float*)alloc((size_t)B * N2 * 128 * 4);
  float* f_feat2 = (float*)alloc((size_t)B * N2 * 256 * 4);
  float* f_int1  = (float*)alloc((size_t)B * N1 * 256 * 4);
  float* f_fu1   = (float*)alloc((size_t)B * N1 * 128 * 4);
  float* f_fu1b  = (float*)alloc((size_t)B * N1 * 128 * 4);
  float* f_int0  = (float*)alloc((size_t)B * N * 128 * 4);
  float* f_fu0   = (float*)alloc((size_t)B * N * 64 * 4);
  float* f_fu0b  = (float*)alloc((size_t)B * N * 64 * 4);
  float* f_latp  = (float*)alloc((size_t)B * N * 128 * 4);
  float* f_lat   = (float*)alloc((size_t)B * N * 128 * 4);
  float* f_nrm   = (float*)alloc((size_t)B * N * 3 * 4);
  float* f_Wv    = (float*)alloc((size_t)B * N * KNN_K * 4);
  float* f_dis   = (float*)alloc((size_t)B * N * 4);
  float* f_cs    = (float*)alloc((size_t)B * (N / 2) * 2 * 4);
  int*   i_perm  = (int*)alloc((size_t)B * N * 4);
  __bf16* bfA = (__bf16*)alloc((size_t)13 * 1024 * 1024 * 2);  // max 131072 x 96 bf16
  __bf16* bfB = (__bf16*)alloc((size_t)65536 * 2);
  float* f_tmpE = (float*)alloc((size_t)B * N * KNN_K * 64 * 4); // 131072 x 64 f32 cap
  float* f_tmpS = (float*)alloc((size_t)B * N * 64 * 4);
  float* f_A = (float*)alloc((size_t)UN * 4);
  float* f_L = (float*)alloc((size_t)UN * 4);
  float* f_U = (float*)alloc((size_t)UN * 4);

  // ---- encoder ----
  sb_knn<<<nblk((long long)B * N), 256, 0, stream>>>(xyz, i_idx0, B, N);
  sb_initfeat<<<nblk((long long)B * N * 4), 256, 0, stream>>>(f_feat, B * N);
  run_edge_conv(stream, f_feat, xyz, i_idx0, w_self0, w_edge0, f_feat0,
                B, N, 4, 64, bfA, bfB, f_tmpE, f_tmpS);

  sb_fps<<<B, 256, 0, stream>>>(xyz, i_fps1, N, N1);
  sb_gather<<<nblk((long long)B * N1 * 3), 256, 0, stream>>>(xyz, i_fps1, f_pos1, B, N1, N, 3);
  sb_gather<<<nblk((long long)B * N1 * 64), 256, 0, stream>>>(f_feat0, i_fps1, f_f0g, B, N1, N, 64);
  sb_knn<<<nblk((long long)B * N1), 256, 0, stream>>>(f_pos1, i_idx1, B, N1);
  run_edge_conv(stream, f_f0g, f_pos1, i_idx1, w_self1, w_edge1, f_feat1,
                B, N1, 64, 128, bfA, bfB, f_tmpE, f_tmpS);

  sb_fps<<<B, 256, 0, stream>>>(f_pos1, i_fps2, N1, N2);
  sb_gather<<<nblk((long long)B * N2 * 3), 256, 0, stream>>>(f_pos1, i_fps2, f_pos2, B, N2, N1, 3);
  sb_gather<<<nblk((long long)B * N2 * 128), 256, 0, stream>>>(f_feat1, i_fps2, f_f1g, B, N2, N1, 128);
  sb_knn<<<nblk((long long)B * N2), 256, 0, stream>>>(f_pos2, i_idx2, B, N2);
  run_edge_conv(stream, f_f1g, f_pos2, i_idx2, w_self2, w_edge2, f_feat2,
                B, N2, 128, 256, bfA, bfB, f_tmpE, f_tmpS);

  // ---- decoder ----
  sb_three_nn<<<nblk((long long)B * N1), 256, 0, stream>>>(f_pos1, f_pos2, i_t3, f_w3, B, N1, N2);
  sb_interp<<<nblk((long long)B * N1 * 256), 256, 0, stream>>>(f_feat2, i_t3, f_w3, f_int1, B, N1, N2, 256);
  sb_concat_bf16<<<nblk((long long)B * N1 * 384), 256, 0, stream>>>(f_int1, 256, f_feat1, 128, bfA, B * N1, 384);
  sb_cvt_pad<<<nblk((long long)384 * 128), 256, 0, stream>>>(w_up1, bfB, 384, 128, 384, 128);
  launch_gemm(stream, bfA, bfB, b_up1, f_fu1, B * N1, 128, 384, 1);
  run_edge_conv(stream, f_fu1, f_pos1, i_idx1, w_cu1, w_edge_cu1, f_fu1b,
                B, N1, 128, 128, bfA, bfB, f_tmpE, f_tmpS);

  sb_three_nn<<<nblk((long long)B * N), 256, 0, stream>>>(xyz, f_pos1, i_t3, f_w3, B, N, N1);
  sb_interp<<<nblk((long long)B * N * 128), 256, 0, stream>>>(f_fu1b, i_t3, f_w3, f_int0, B, N, N1, 128);
  sb_concat_bf16<<<nblk((long long)B * N * 192), 256, 0, stream>>>(f_int0, 128, f_feat0, 64, bfA, B * N, 192);
  sb_cvt_pad<<<nblk((long long)192 * 64), 256, 0, stream>>>(w_up0, bfB, 192, 64, 192, 64);
  launch_gemm(stream, bfA, bfB, b_up0, f_fu0, B * N, 64, 192, 1);
  run_edge_conv(stream, f_fu0, xyz, i_idx0, w_cu0, w_edge_cu0, f_fu0b,
                B, N, 64, 64, bfA, bfB, f_tmpE, f_tmpS);

  // latent = l2norm(fu0 @ w_out + b_out)
  sb_cvt_pad<<<nblk((long long)B * N * 64), 256, 0, stream>>>(f_fu0b, bfA, B * N, 64, B * N, 64);
  sb_cvt_pad<<<nblk((long long)64 * 128), 256, 0, stream>>>(w_out, bfB, 64, 128, 64, 128);
  launch_gemm(stream, bfA, bfB, b_out, f_latp, B * N, 128, 64, 2);
  sb_l2norm<<<nblk((long long)B * N), 256, 0, stream>>>(f_latp, f_lat, B * N, 128);

  // ---- edge weights + geometric loss + idx output ----
  sb_wval<<<nblk((long long)B * N * KNN_K), 256, 0, stream>>>(f_lat, i_idx0, f_Wv, B, N, 128);
  sb_normals<<<nblk((long long)B * N), 256, 0, stream>>>(ps, cidx, f_nrm, B, N, Hh * Ww);
  sb_zero<<<1, 256, 0, stream>>>(out_loss, 1);
  sb_loss<<<nblk((long long)B * N * KNN_K), 256, 0, stream>>>(f_Wv, f_nrm, i_idx0, alpha, out_loss, B, N);
  sb_idx_copy<<<nblk((long long)B * N * KNN_K), 256, 0, stream>>>(out_idx, i_idx0, B * N * KNN_K);

  // ---- normalized Laplacian ----
  hipMemsetAsync(f_A, 0, (size_t)UN * 4, stream);
  sb_scatter<<<nblk((long long)B * N * KNN_K), 256, 0, stream>>>(f_Wv, i_idx0, f_A, B, N);
  sb_degree<<<nblk((long long)B * N), 256, 0, stream>>>(f_A, f_dis, B, N);
  sb_buildL<<<nblk(UN), 256, 0, stream>>>(f_A, f_dis, f_L, B, N);
  sb_initU<<<nblk(UN), 256, 0, stream>>>(f_U, B, N);

  // ---- parallel cyclic Jacobi (L,U fit in the 192MB L2) ----
  const int SWEEPS = 4;
  const long long PW = (long long)B * (N / 2) * N;
  for (int sw = 0; sw < SWEEPS; ++sw) {
    for (int st = 0; st < N - 1; ++st) {
      sb_jrot<<<nblk((long long)B * (N / 2)), 256, 0, stream>>>(f_L, f_cs, st, B, N);
      sb_jrows<<<nblk(PW), 256, 0, stream>>>(f_L, f_cs, st, B, N);
      sb_jcols<<<nblk(PW), 256, 0, stream>>>(f_L, f_U, f_cs, st, B, N);
    }
  }

  // ---- ascending eigenvalue order, write U to output ----
  sb_sortperm<<<B, 1024, 0, stream>>>(f_L, i_perm, N);
  sb_permU<<<nblk(UN), 256, 0, stream>>>(f_U, i_perm, out_U, B, N);
}
